// CausalSlidingWindowGQA_7464653160668
// MI455X (gfx1250) — matrix operand order repaired and checked by reading, not compiled, not run on hardware
//
#include <hip/hip_runtime.h>
#include <hip/hip_bf16.h>

// ---------------------------------------------------------------------------
// CDNA5 (gfx1250) WMMA implementation of CausalSlidingWindowGQA.
// - All matmuls: v_wmma_f32_16x16x32_f16 (f16 in, f32 accumulate).
// - Frame-granular causality + sliding window => only full 32-key blocks:
//   past keys 3072..4095 plus new keys 0..(qframe+1)*64-1. No masking.
// - K/V tiles staged into LDS by the Tensor Data Mover (TENSOR_LOAD_TO_LDS),
//   double-buffered, completion via s_wait_tensorcnt.
// ---------------------------------------------------------------------------

typedef _Float16 f16;
typedef _Float16 h8_t  __attribute__((ext_vector_type(8)));
typedef _Float16 v16h  __attribute__((ext_vector_type(16)));
typedef float    v8f   __attribute__((ext_vector_type(8)));
typedef unsigned int u32x4 __attribute__((ext_vector_type(4)));
typedef int          i32x4 __attribute__((ext_vector_type(4)));
typedef int          i32x8 __attribute__((ext_vector_type(8)));

#if __has_builtin(__builtin_amdgcn_tensor_load_to_lds) && \
    __has_builtin(__builtin_amdgcn_s_wait_tensorcnt)
#define HAVE_TDM 1
#else
#define HAVE_TDM 0
#endif

// A-fragment (16x32 f16, M x K), src row-major with leading dim ld.
// Lane layout: row = lane%16; lane<16 holds K 0-7 & 16-23, lane>=16 holds
// K 8-15 & 24-31 (ISA 7.12.2, 16-bit A 16x32).
__device__ __forceinline__ v16h load_a_frag(const f16* src, int ld) {
  int l   = threadIdx.x & 31;
  int row = l & 15;
  int khi = l >> 4;
  const f16* p = src + row * ld + khi * 8;
  h8_t lo = *(const h8_t*)(p);
  h8_t hi = *(const h8_t*)(p + 16);
  v16h out;
#pragma unroll
  for (int i = 0; i < 8; ++i) { out[i] = lo[i]; out[i + 8] = hi[i]; }
  return out;
}

// B-fragment (32x16 f16, K x N) from a transposed (N x K row-major) source.
// Lane layout: col = lane%16; lane<16 holds K 0-7 & 16-23, lane>=16 holds K 8-15 & 24-31.
__device__ __forceinline__ v16h load_b_fragT(const f16* bt, int ld) {
  int l   = threadIdx.x & 31;
  int col = l & 15;
  int kh  = l >> 4;
  const f16* p = bt + col * ld + kh * 8;
  h8_t lo = *(const h8_t*)(p);
  h8_t hi = *(const h8_t*)(p + 16);
  v16h out;
#pragma unroll
  for (int i = 0; i < 8; ++i) { out[i] = lo[i]; out[i + 8] = hi[i]; }
  return out;
}

__device__ __forceinline__ v8f wmma_f16(v16h a, v16h b, v8f c) {
  return __builtin_amdgcn_wmma_f32_16x16x32_f16(
      /*neg_a=*/false, a, /*neg_b=*/false, b,
      /*c_mod=*/(short)0, c, /*reuse_a=*/false, /*reuse_b=*/false);
}

// Generic LDS pointer -> 32-bit LDS byte offset (LDS generic addresses carry
// the offset in the low 32 bits; aperture in the high bits).
__device__ __forceinline__ unsigned lds_off(const void* p) {
  return (unsigned)(unsigned long long)p;
}

#if HAVE_TDM
// Issue a 2D TDM tile load (f16 elements): tile_d1 rows of tile_d0 elements,
// row stride stride_d0 elements, into LDS at lds_byte_off.
// D# group0: count=1 | lds_addr | global_addr(57b) | type=2.
// D# group1: data_size=1(2B) | tensor dims | tile dims | dim0 stride.
__device__ __forceinline__ void tdm_load_2d_f16(
    unsigned lds_byte_off, const f16* gptr,
    unsigned tensor_d0, unsigned tensor_d1,
    unsigned tile_d0, unsigned tile_d1, unsigned stride_d0) {
  unsigned long long ga = (unsigned long long)gptr;
  u32x4 g0;
  g0[0] = 1u;                                        // count=1, user mode
  g0[1] = lds_byte_off;                              // lds_addr
  g0[2] = (unsigned)(ga & 0xffffffffu);              // global_addr[31:0]
  g0[3] = (unsigned)((ga >> 32) & 0x01ffffffu)       // global_addr[56:32]
        | (2u << 30);                                // type = 2 ("image")
  i32x8 g1;
  g1[0] = (int)(1u << 16);                           // data_size = 1 (2 bytes)
  g1[1] = (int)((tensor_d0 & 0xffffu) << 16);        // tensor_dim0[15:0]
  g1[2] = (int)((tensor_d0 >> 16) |
                ((tensor_d1 & 0xffffu) << 16));      // dim0[31:16] | dim1[15:0]
  g1[3] = (int)((tensor_d1 >> 16) |
                ((tile_d0 & 0xffffu) << 16));        // dim1[31:16] | tile_dim0
  g1[4] = (int)(tile_d1 & 0xffffu);                  // tile_dim1 (tile_dim2=0)
  g1[5] = (int)stride_d0;                            // tensor_dim0_stride[31:0]
  g1[6] = 0;
  g1[7] = 0;
  i32x4 z4 = {0, 0, 0, 0};
#if __clang_major__ >= 23
  i32x8 z8 = {0, 0, 0, 0, 0, 0, 0, 0};
  __builtin_amdgcn_tensor_load_to_lds(g0, g1, z4, z4, z8, 0);
#else
  __builtin_amdgcn_tensor_load_to_lds(g0, g1, z4, z4, 0);
#endif
}
#endif  // HAVE_TDM

// ---------------------------------------------------------------------------
// Conversions
// ---------------------------------------------------------------------------
__global__ void f32_to_f16_kernel(const float* __restrict__ src,
                                  f16* __restrict__ dst, int n) {
  int i = blockIdx.x * blockDim.x + threadIdx.x;
  if (i < n) dst[i] = (f16)src[i];
}

// W: K x N f32  ->  WT: N x K f16
__global__ void transpose_f32_to_f16_kernel(const float* __restrict__ W,
                                            f16* __restrict__ WT, int K, int N) {
  int i = blockIdx.x * blockDim.x + threadIdx.x;
  if (i >= K * N) return;
  int k = i / N, n = i - k * N;
  WT[(size_t)n * K + k] = (f16)W[i];
}

// ---------------------------------------------------------------------------
// WMMA GEMM: C[M,N] = A[M,K] * B[K,N], B given transposed (N x K row-major).
// Block = 4 waves; wave computes a 16x64 stripe of a 64x64 tile.
// Software-pipelined: iteration i+1's fragments load while i's WMMAs run.
// ---------------------------------------------------------------------------
__global__ __launch_bounds__(128) void gemm_f16_tn(
    const f16* __restrict__ A, const f16* __restrict__ BT,
    float* __restrict__ C, int M, int N, int K) {
  int wave = threadIdx.x >> 5;
  int row0 = blockIdx.y * 64 + wave * 16;
  int col0 = blockIdx.x * 64;
  const f16* aP  = A + (size_t)row0 * K;
  const f16* bP0 = BT + (size_t)(col0 + 0)  * K;
  const f16* bP1 = BT + (size_t)(col0 + 16) * K;
  const f16* bP2 = BT + (size_t)(col0 + 32) * K;
  const f16* bP3 = BT + (size_t)(col0 + 48) * K;

  v8f acc[4];
#pragma unroll
  for (int n = 0; n < 4; ++n)
#pragma unroll
    for (int j = 0; j < 8; ++j) acc[n][j] = 0.f;

  v16h a  = load_a_frag(aP, K);
  v16h b0 = load_b_fragT(bP0, K);
  v16h b1 = load_b_fragT(bP1, K);
  v16h b2 = load_b_fragT(bP2, K);
  v16h b3 = load_b_fragT(bP3, K);

  for (int k0 = 32; k0 < K; k0 += 32) {
    v16h an  = load_a_frag(aP + k0, K);
    v16h bn0 = load_b_fragT(bP0 + k0, K);
    v16h bn1 = load_b_fragT(bP1 + k0, K);
    v16h bn2 = load_b_fragT(bP2 + k0, K);
    v16h bn3 = load_b_fragT(bP3 + k0, K);
    acc[0] = wmma_f16(a, b0, acc[0]);
    acc[1] = wmma_f16(a, b1, acc[1]);
    acc[2] = wmma_f16(a, b2, acc[2]);
    acc[3] = wmma_f16(a, b3, acc[3]);
    a = an; b0 = bn0; b1 = bn1; b2 = bn2; b3 = bn3;
  }
  acc[0] = wmma_f16(a, b0, acc[0]);
  acc[1] = wmma_f16(a, b1, acc[1]);
  acc[2] = wmma_f16(a, b2, acc[2]);
  acc[3] = wmma_f16(a, b3, acc[3]);

  int colo = threadIdx.x & 15;
  int half = (threadIdx.x >> 4) & 1;
#pragma unroll
  for (int n = 0; n < 4; ++n)
#pragma unroll
    for (int j = 0; j < 8; ++j)
      C[(size_t)(row0 + j + 8 * half) * N + col0 + n * 16 + colo] = acc[n][j];
}

// ---------------------------------------------------------------------------
// RoPE: angle = inv_freq_i * (t + 100*(d + b)); new tokens: t = q/64 + 64.
// ---------------------------------------------------------------------------
__device__ __forceinline__ void rope_angle(int tok, int pair, float* s, float* c) {
  float invf = powf(10000.0f, -(float)pair * (1.0f / 32.0f));
  float t  = (float)(tok >> 6) + 64.0f;
  float dp = (float)((tok & 63) >> 3);
  float bp = (float)(tok & 7);
  float ang = invf * (t + 100.0f * (dp + bp));
  *s = sinf(ang);
  *c = cosf(ang);
}

// Q rope: Qf32[q][h*64+d] -> Qh[h][q][d] (f16), with 1/sqrt(64) folded in.
__global__ void rope_q_kernel(const float* __restrict__ Qf, f16* __restrict__ Qh) {
  int idx  = blockIdx.x * blockDim.x + threadIdx.x;   // 1024*16*32
  int pair = idx & 31;
  int h    = (idx >> 5) & 15;
  int q    = idx >> 9;
  float s, c;
  rope_angle(q, pair, &s, &c);
  const float* src = Qf + (size_t)q * 1024 + h * 64 + 2 * pair;
  float x1 = src[0], x2 = src[1];
  float y1 = x1 * c - x2 * s;
  float y2 = x1 * s + x2 * c;
  f16* dst = Qh + ((size_t)h * 1024 + q) * 64 + 2 * pair;
  dst[0] = (f16)(y1 * 0.125f);
  dst[1] = (f16)(y2 * 0.125f);
}

// Kn rope + Vn: emit f32 outputs (d_out regions) and f16 packed K/V.
// Khp: [4][2048][64]  (slot 1024+j = new key j); VhT: [4][64][2048].
__global__ void rope_kv_kernel(const float* __restrict__ Knf,
                               const float* __restrict__ Vnf,
                               float* __restrict__ out_kn,
                               float* __restrict__ out_vn,
                               f16* __restrict__ Khp, f16* __restrict__ VhT) {
  int idx  = blockIdx.x * blockDim.x + threadIdx.x;   // 1024*4*32
  int pair = idx & 31;
  int g    = (idx >> 5) & 3;
  int j    = idx >> 7;
  float s, c;
  rope_angle(j, pair, &s, &c);
  float k1 = Knf[(size_t)j * 256 + g * 64 + 2 * pair];
  float k2 = Knf[(size_t)j * 256 + g * 64 + 2 * pair + 1];
  float y1 = k1 * c - k2 * s;
  float y2 = k1 * s + k2 * c;
  float* ok = out_kn + ((size_t)g * 1024 + j) * 64 + 2 * pair;
  ok[0] = y1; ok[1] = y2;
  f16* kh = Khp + ((size_t)g * 2048 + 1024 + j) * 64 + 2 * pair;
  kh[0] = (f16)y1; kh[1] = (f16)y2;
  float v1 = Vnf[(size_t)j * 256 + g * 64 + 2 * pair];
  float v2 = Vnf[(size_t)j * 256 + g * 64 + 2 * pair + 1];
  float* ov = out_vn + ((size_t)g * 1024 + j) * 64 + 2 * pair;
  ov[0] = v1; ov[1] = v2;
  VhT[((size_t)g * 64 + 2 * pair) * 2048 + 1024 + j]     = (f16)v1;
  VhT[((size_t)g * 64 + 2 * pair + 1) * 2048 + 1024 + j] = (f16)v2;
}

// Windowed past: only keys 3072..4095 can ever be attended (t_max - 31 = 48).
__global__ void past_pack_kernel(const float* __restrict__ pk,
                                 const float* __restrict__ pv,
                                 f16* __restrict__ Khp, f16* __restrict__ VhT) {
  int idx = blockIdx.x * blockDim.x + threadIdx.x;    // 4*1024*64
  int d = idx & 63;
  int p = (idx >> 6) & 1023;
  int g = idx >> 16;
  size_t src = ((size_t)g * 4096 + 3072 + p) * 64 + d;
  Khp[((size_t)g * 2048 + p) * 64 + d]  = (f16)pk[src];
  VhT[((size_t)g * 64 + d) * 2048 + p]  = (f16)pv[src];
}

// ---------------------------------------------------------------------------
// Flash attention: grid (head, qframe), 4 waves; wave owns 16 q rows.
// nblocks = 34 + 2*qf full 32-key blocks; K/V staged by TDM, double-buffered.
// ---------------------------------------------------------------------------
__global__ __launch_bounds__(128) void attn_kernel(
    const f16* __restrict__ Qh, const f16* __restrict__ Khp,
    const f16* __restrict__ VhT, f16* __restrict__ attnH) {
  __shared__ alignas(16) f16 Ksh[2][32 * 64];   // [buf][key][dim]
  __shared__ alignas(16) f16 Vsh[2][64 * 32];   // [buf][dim][key] (transposed)
  __shared__ alignas(16) f16 Psh[4][16 * 32];   // per-wave exp(S) tile

  int h    = blockIdx.x;
  int qf   = blockIdx.y;
  int g    = h >> 2;                    // GQA: rep = 4
  int wave = threadIdx.x >> 5;
  int lane = threadIdx.x & 31;
  int colo = lane & 15;
  int half = lane >> 4;
  int q0   = qf * 64 + wave * 16;

  const f16* Qb = Qh + ((size_t)h * 1024 + q0) * 64;
  v16h aq0 = load_a_frag(Qb, 64);        // dims 0..31 (scale prefolded)
  v16h aq1 = load_a_frag(Qb + 32, 64);   // dims 32..63

  v8f o[4];
  float m[8], lsum[8], alpha[8];
#pragma unroll
  for (int n = 0; n < 4; ++n)
#pragma unroll
    for (int j = 0; j < 8; ++j) o[n][j] = 0.f;
#pragma unroll
  for (int j = 0; j < 8; ++j) { m[j] = -3.0e38f; lsum[j] = 0.f; }

  const f16* Kg = Khp + (size_t)g * 2048 * 64;
  const f16* Vg = VhT + (size_t)g * 64 * 2048;
  int nblocks = 34 + 2 * qf;

#if HAVE_TDM
  bool issuer = (threadIdx.x < 32);     // wave 0 drives the TDM
  if (issuer) {
    tdm_load_2d_f16(lds_off(Ksh[0]), Kg,      64, 2048, 64, 32, 64);
    tdm_load_2d_f16(lds_off(Vsh[0]), Vg,    2048,   64, 32, 64, 2048);
  }
#endif

  for (int b = 0; b < nblocks; ++b) {
    int cur = b & 1;
#if HAVE_TDM
    if (issuer) __builtin_amdgcn_s_wait_tensorcnt(0);
    __syncthreads();                    // data for block b visible to all waves
    if (issuer && (b + 1) < nblocks) {  // prefetch block b+1 into other buffer
      int k1 = (b + 1) * 32;
      tdm_load_2d_f16(lds_off(Ksh[cur ^ 1]), Kg + (size_t)k1 * 64,
                      64, 2048, 64, 32, 64);
      tdm_load_2d_f16(lds_off(Vsh[cur ^ 1]), Vg + k1,
                      2048, 64, 32, 64, 2048);
    }
#else
    __syncthreads();                    // prior reads of buf done
    {
      int key0 = b * 32;
      int tid = threadIdx.x;
      const h8_t* ks = (const h8_t*)(Kg + (size_t)key0 * 64);
      h8_t* kd = (h8_t*)Ksh[cur];
      kd[tid * 2]     = ks[tid * 2];
      kd[tid * 2 + 1] = ks[tid * 2 + 1];
      int d = tid >> 1, hr = tid & 1;
      const h8_t* vs = (const h8_t*)(Vg + (size_t)d * 2048 + key0 + hr * 16);
      h8_t* vd = (h8_t*)(Vsh[cur] + d * 32 + hr * 16);
      vd[0] = vs[0];
      vd[1] = vs[1];
    }
    __syncthreads();
#endif
    const f16* Kb = Ksh[cur];
    const f16* Vb = Vsh[cur];

    // S = Q K^T for two 16-key column tiles
    v8f s0, s1;
#pragma unroll
    for (int j = 0; j < 8; ++j) { s0[j] = 0.f; s1[j] = 0.f; }
    {
      v16h bk0 = load_b_fragT(Kb, 64);
      v16h bk1 = load_b_fragT(Kb + 16 * 64, 64);
      v16h bk2 = load_b_fragT(Kb + 32, 64);
      v16h bk3 = load_b_fragT(Kb + 16 * 64 + 32, 64);
      s0 = wmma_f16(aq0, bk0, s0);
      s1 = wmma_f16(aq0, bk1, s1);
      s0 = wmma_f16(aq1, bk2, s0);
      s1 = wmma_f16(aq1, bk3, s1);
    }

    // Online softmax. C-frag slot j holds row (j + 8*half), col = lane%16;
    // xor masks 1..8 reduce within each 16-lane half (matches row layout).
    f16* Pw = Psh[wave];
#pragma unroll
    for (int j = 0; j < 8; ++j) {
      float v = fmaxf(s0[j], s1[j]);
#pragma unroll
      for (int off = 8; off > 0; off >>= 1) v = fmaxf(v, __shfl_xor(v, off, 32));
      float mn = fmaxf(m[j], v);
      float al = __expf(m[j] - mn);
      float p0 = __expf(s0[j] - mn);
      float p1 = __expf(s1[j] - mn);
      float rs = p0 + p1;
#pragma unroll
      for (int off = 8; off > 0; off >>= 1) rs += __shfl_xor(rs, off, 32);
      lsum[j] = lsum[j] * al + rs;
      m[j]    = mn;
      alpha[j] = al;
      int row = j + 8 * half;
      Pw[row * 32 + colo]      = (f16)p0;
      Pw[row * 32 + 16 + colo] = (f16)p1;
    }

    // O = alpha*O + P V  (one 16x32 A-frag covers the whole 32-key block)
    v16h ap = load_a_frag(Pw, 32);
#pragma unroll
    for (int n = 0; n < 4; ++n) {
#pragma unroll
      for (int j = 0; j < 8; ++j) o[n][j] *= alpha[j];
      v16h bv = load_b_fragT(Vb + n * 16 * 32, 32);
      o[n] = wmma_f16(ap, bv, o[n]);
    }
  }

  // epilogue: normalize and write f16 [token][h*64+d] for the out-projection
#pragma unroll
  for (int n = 0; n < 4; ++n)
#pragma unroll
    for (int j = 0; j < 8; ++j) {
      int row = q0 + j + 8 * half;
      float val = o[n][j] / lsum[j];
      attnH[(size_t)row * 1024 + h * 64 + n * 16 + colo] = (f16)val;
    }
}

// ---------------------------------------------------------------------------
extern "C" void kernel_launch(void* const* d_in, const int* in_sizes, int n_in,
                              void* d_out, int out_size, void* d_ws, size_t ws_size,
                              hipStream_t stream) {
  (void)in_sizes; (void)n_in; (void)out_size; (void)ws_size;
  const float* hidden = (const float*)d_in[0];
  const float* past_k = (const float*)d_in[1];
  const float* past_v = (const float*)d_in[2];
  const float* wq     = (const float*)d_in[3];
  const float* wk     = (const float*)d_in[4];
  const float* wv     = (const float*)d_in[5];
  const float* wo     = (const float*)d_in[6];
  float* out = (float*)d_out;
  float* out_kn = out + 1024 * 1024;            // (1,4,1024,64)
  float* out_vn = out_kn + 4 * 1024 * 64;       // (1,4,1024,64)

  char* ws = (char*)d_ws;
  size_t off = 0;
  auto alloc = [&](size_t bytes) {
    char* p = ws + off;
    off += (bytes + 255) & ~(size_t)255;
    return p;
  };
  f16*   Ah    = (f16*)alloc((size_t)1024 * 1024 * 2);
  f16*   WqT   = (f16*)alloc((size_t)1024 * 1024 * 2);
  f16*   WkT   = (f16*)alloc((size_t)256 * 1024 * 2);
  f16*   WvT   = (f16*)alloc((size_t)256 * 1024 * 2);
  f16*   WoT   = (f16*)alloc((size_t)1024 * 1024 * 2);
  float* Qf32  = (float*)alloc((size_t)1024 * 1024 * 4);
  float* Knf32 = (float*)alloc((size_t)1024 * 256 * 4);
  float* Vnf32 = (float*)alloc((size_t)1024 * 256 * 4);
  f16*   Qh    = (f16*)alloc((size_t)16 * 1024 * 64 * 2);
  f16*   Khp   = (f16*)alloc((size_t)4 * 2048 * 64 * 2);
  f16*   VhT   = (f16*)alloc((size_t)4 * 64 * 2048 * 2);
  f16*   attnH = (f16*)alloc((size_t)1024 * 1024 * 2);

  // 1) conversions / weight transposes
  f32_to_f16_kernel<<<(1024 * 1024) / 256, 256, 0, stream>>>(hidden, Ah, 1024 * 1024);
  transpose_f32_to_f16_kernel<<<(1024 * 1024) / 256, 256, 0, stream>>>(wq, WqT, 1024, 1024);
  transpose_f32_to_f16_kernel<<<(1024 * 256) / 256, 256, 0, stream>>>(wk, WkT, 1024, 256);
  transpose_f32_to_f16_kernel<<<(1024 * 256) / 256, 256, 0, stream>>>(wv, WvT, 1024, 256);
  transpose_f32_to_f16_kernel<<<(1024 * 1024) / 256, 256, 0, stream>>>(wo, WoT, 1024, 1024);

  // 2) QKV projections (WMMA)
  gemm_f16_tn<<<dim3(1024 / 64, 1024 / 64), 128, 0, stream>>>(Ah, WqT, Qf32, 1024, 1024, 1024);
  gemm_f16_tn<<<dim3(256 / 64, 1024 / 64), 128, 0, stream>>>(Ah, WkT, Knf32, 1024, 256, 1024);
  gemm_f16_tn<<<dim3(256 / 64, 1024 / 64), 128, 0, stream>>>(Ah, WvT, Vnf32, 1024, 256, 1024);

  // 3) RoPE + packing (also emits Kn/Vn f32 outputs)
  rope_q_kernel<<<(1024 * 16 * 32) / 256, 256, 0, stream>>>(Qf32, Qh);
  rope_kv_kernel<<<(1024 * 4 * 32) / 256, 256, 0, stream>>>(Knf32, Vnf32, out_kn, out_vn, Khp, VhT);
  past_pack_kernel<<<(4 * 1024 * 64) / 256, 256, 0, stream>>>(past_k, past_v, Khp, VhT);

  // 4) flash attention (WMMA + TDM double-buffered staging)
  attn_kernel<<<dim3(16, 16), 128, 0, stream>>>(Qh, Khp, VhT, attnH);

  // 5) output projection (WMMA) -> d_out
  gemm_f16_tn<<<dim3(1024 / 64, 1024 / 64), 128, 0, stream>>>(attnH, WoT, out, 1024, 1024, 1024);
}
